// EdgeDecoder_88364657148159
// MI455X (gfx1250) — matrix operand order repaired and checked by reading, not compiled, not run on hardware
//
#include <hip/hip_runtime.h>
#include <cstdint>

// EdgeDecoder for MI455X (gfx1250, wave32).
//  - z is pre-converted to bf16 into d_ws (51MB, L2-resident): halves the
//    dominant L2 gather traffic (2KB -> 1KB per edge) and lets the A-fragment
//    build use v_pk_mul_bf16 (inline asm) instead of f32 mul + cvt.
//  - W1 transposed+bf16-converted once into d_ws, then DMA'd into LDS
//    (128KB of the 320KB WGP LDS) per block via TENSOR_LOAD_TO_LDS (TDM).
//  - GEMM: v_wmma_f32_16x16x32_bf16; A-fragments in registers, B-fragments
//    from LDS (ds_load_b128), so DS pipe feeds WMMA while VMEM does gathers.
//  - bias + relu + W2-dot + sigmoid fused into the epilogue; no [E,256]
//    intermediate ever touches HBM.

#define WAVES_PER_BLOCK 8
#define TILE_M 16
#define CH 256                    // IN_CH == HID == 256
#define WT_BYTES (CH * CH * 2)    // 131072

#if defined(__HIP_DEVICE_COMPILE__)
typedef __attribute__((ext_vector_type(16))) __bf16 v16bf;
typedef __attribute__((ext_vector_type(8)))  float  v8f;
typedef __attribute__((ext_vector_type(4)))  unsigned int u32x4;
typedef __attribute__((ext_vector_type(8)))  int          i32x8;
typedef __attribute__((ext_vector_type(4)))  int          i32x4;

__device__ __forceinline__ uint32_t pkmul_bf16(uint32_t a, uint32_t b) {
    uint32_t r;
    asm("v_pk_mul_bf16 %0, %1, %2" : "=v"(r) : "v"(a), "v"(b));
    return r;
}
#endif

// Prep: Wt[n][k] = bf16(W1[k][n]); 128KB total, L2-hot.
__global__ __launch_bounds__(256) void edgedec_prep_w1t(
    const float* __restrict__ W1, unsigned short* __restrict__ Wt)
{
#if defined(__HIP_DEVICE_COMPILE__)
    int idx = blockIdx.x * blockDim.x + threadIdx.x;  // 65536 threads total
    int n = idx >> 8;
    int k = idx & 255;
    __bf16* wt = reinterpret_cast<__bf16*>(Wt);
    wt[n * CH + k] = (__bf16)W1[k * CH + n];
#endif
}

// Prep: zb = bf16(z), streaming convert (4 elems/thread).
__global__ __launch_bounds__(256) void edgedec_prep_zbf(
    const float4* __restrict__ z4, ushort4* __restrict__ zb4, int n4)
{
#if defined(__HIP_DEVICE_COMPILE__)
    int i = blockIdx.x * blockDim.x + threadIdx.x;
    if (i < n4) {
        float4 f = z4[i];
        union { __bf16 b[4]; ushort4 s; } c;
        c.b[0] = (__bf16)f.x; c.b[1] = (__bf16)f.y;
        c.b[2] = (__bf16)f.z; c.b[3] = (__bf16)f.w;
        zb4[i] = c.s;
    }
#endif
}

template <bool ZBF>
__global__ __launch_bounds__(256) void edgedec_main(
    const float* __restrict__ z,              // fp32 z (fallback path)
    const unsigned short* __restrict__ zb,    // bf16 z (fast path)
    const long long* __restrict__ src,        // reference dtype is int64
    const long long* __restrict__ dst,
    const unsigned short* __restrict__ Wt_,   // bf16 [HID][IN_CH] (n-major)
    const float* __restrict__ b1,
    const float* __restrict__ W2,
    const float* __restrict__ b2,
    float* __restrict__ out,
    int E)
{
#if defined(__HIP_DEVICE_COMPILE__)
    extern __shared__ unsigned short smem[];  // 131072B dynamic LDS (offset 0)
    const int lane = threadIdx.x & 31;
    const int wave = threadIdx.x >> 5;
    const int row  = lane & 15;   // A-matrix row M = lane % 16
    const int half = lane >> 4;   // lane half selects K sub-ranges

    // ---- Stage Wt (bf16 [256][256], 128KB) into LDS via the TDM ----
    if (wave == 0) {
        uint64_t ga       = (uint64_t)reinterpret_cast<uintptr_t>(Wt_);
        uint32_t lds_base = (uint32_t)reinterpret_cast<uintptr_t>(&smem[0]);
        // D# group 0: count=1 | lds_addr | global_addr[56:0] | type=2
        u32x4 g0 = { 0x1u,
                     lds_base,
                     (uint32_t)ga,
                     ((uint32_t)(ga >> 32) & 0x01FFFFFFu) | (2u << 30) };
        // D# group 1: 2D tensor, data_size=2B, dims 256x256, stride0=256
        i32x8 g1 = { (int)(1u << 16),     // data_size=1 (2 bytes); mask/pad=0
                     (int)(256u << 16),   // [63:48]  tensor_dim0 = 256
                     (int)(256u << 16),   // [95:80]  tensor_dim1 = 256
                     (int)(256u << 16),   // [127:112] tile_dim0  = 256
                     (int)256,            // [143:128] tile_dim1  = 256
                     (int)256,            // [191:160] tensor_dim0_stride = 256
                     0, 0 };
        i32x4 g2 = { 0, 0, 0, 0 };
        i32x4 g3 = { 0, 0, 0, 0 };
#if defined(__clang_major__) && (__clang_major__ >= 23)
        i32x8 g4 = { 0, 0, 0, 0, 0, 0, 0, 0 };
        __builtin_amdgcn_tensor_load_to_lds(g0, g1, g2, g3, g4, 0);
#else
        __builtin_amdgcn_tensor_load_to_lds(g0, g1, g2, g3, 0);
#endif
        __builtin_amdgcn_s_wait_tensorcnt(0);
    }
    __syncthreads();
    const __bf16* Wl = reinterpret_cast<const __bf16*>(smem);

    // ---- Hoist edge-invariant epilogue constants (b1, W2) ----
    float b1r[16], w2r[16];
#pragma unroll
    for (int nt = 0; nt < 16; ++nt) {
        b1r[nt] = b1[nt * 16 + row];
        w2r[nt] = W2[nt * 16 + row];
    }
    const float b2v = b2[0];

    const int nTiles = (E + TILE_M - 1) / TILE_M;
    for (int tile = blockIdx.x * WAVES_PER_BLOCK + wave; tile < nTiles;
         tile += (int)gridDim.x * WAVES_PER_BLOCK) {

        int e  = tile * TILE_M + row;
        int eg = e < E ? e : (E - 1);

        // ---- Build 8 bf16 A-fragments (16x32 each) in contiguous K order ----
        // lanes 0-15 : row M=lane,    elems 0..15 = K kb..kb+15
        // lanes 16-31: row M=lane-16, same with kb = kc*32 + 16
        v16bf A[8];
        if constexpr (ZBF) {
            const uint32_t* zs = (const uint32_t*)(zb + (size_t)src[eg] * CH);
            const uint32_t* zd = (const uint32_t*)(zb + (size_t)dst[eg] * CH);
#pragma unroll
            for (int kc = 0; kc < 8; ++kc) {
                int kw = kc * 16 + half * 8;       // u32 word index (= kb/2)
                uint4 sa = *(const uint4*)(zs + kw);
                uint4 sb = *(const uint4*)(zs + kw + 4);
                uint4 da = *(const uint4*)(zd + kw);
                uint4 db = *(const uint4*)(zd + kw + 4);
                union { v16bf v; uint32_t u[8]; } au;
                au.u[0] = pkmul_bf16(sa.x, da.x);
                au.u[1] = pkmul_bf16(sa.y, da.y);
                au.u[2] = pkmul_bf16(sa.z, da.z);
                au.u[3] = pkmul_bf16(sa.w, da.w);
                au.u[4] = pkmul_bf16(sb.x, db.x);
                au.u[5] = pkmul_bf16(sb.y, db.y);
                au.u[6] = pkmul_bf16(sb.z, db.z);
                au.u[7] = pkmul_bf16(sb.w, db.w);
                A[kc] = au.v;
            }
        } else {
            const float* zs = z + (size_t)src[eg] * CH;
            const float* zd = z + (size_t)dst[eg] * CH;
#pragma unroll
            for (int kc = 0; kc < 8; ++kc) {
                int kb = kc * 32 + half * 16;
                float4 s0 = *(const float4*)(zs + kb);
                float4 s1 = *(const float4*)(zs + kb + 4);
                float4 s2 = *(const float4*)(zs + kb + 8);
                float4 s3 = *(const float4*)(zs + kb + 12);
                float4 d0 = *(const float4*)(zd + kb);
                float4 d1 = *(const float4*)(zd + kb + 4);
                float4 d2 = *(const float4*)(zd + kb + 8);
                float4 d3 = *(const float4*)(zd + kb + 12);
                v16bf a;
                a[0]  = (__bf16)(s0.x * d0.x); a[1]  = (__bf16)(s0.y * d0.y);
                a[2]  = (__bf16)(s0.z * d0.z); a[3]  = (__bf16)(s0.w * d0.w);
                a[4]  = (__bf16)(s1.x * d1.x); a[5]  = (__bf16)(s1.y * d1.y);
                a[6]  = (__bf16)(s1.z * d1.z); a[7]  = (__bf16)(s1.w * d1.w);
                a[8]  = (__bf16)(s2.x * d2.x); a[9]  = (__bf16)(s2.y * d2.y);
                a[10] = (__bf16)(s2.z * d2.z); a[11] = (__bf16)(s2.w * d2.w);
                a[12] = (__bf16)(s3.x * d3.x); a[13] = (__bf16)(s3.y * d3.y);
                a[14] = (__bf16)(s3.z * d3.z); a[15] = (__bf16)(s3.w * d3.w);
                A[kc] = a;
            }
        }

        float outacc[8] = {0.f, 0.f, 0.f, 0.f, 0.f, 0.f, 0.f, 0.f};

        // ---- 16 N-tiles of h = relu(x@W1+b1), fused with h@W2 ----
        for (int nt = 0; nt < 16; ++nt) {
            int n = nt * 16 + row;  // B column for this lane
            // B-fragment from LDS: lane holds 16 contiguous K for column n;
            // lane half selects K offset (0 / 16) within the 32-K chunk.
            const __bf16* bp = Wl + n * CH + half * 16;
            v8f acc = {};
#pragma unroll
            for (int kc = 0; kc < 8; ++kc) {
                v16bf B = *(const v16bf*)(bp + kc * 32);  // 2x ds_load_b128
                acc = __builtin_amdgcn_wmma_f32_16x16x32_bf16(
                        false, A[kc], false, B, (short)0, acc, false, false);
            }
            // D layout: lane L: N = L%16; VGPR v: M = v + 8*(L>=16)
            float b1v = b1r[nt];
            float w2v = w2r[nt];
#pragma unroll
            for (int v = 0; v < 8; ++v) {
                float h = acc[v] + b1v;
                h = h > 0.f ? h : 0.f;
                outacc[v] += h * w2v;   // partial dot over this lane's n
            }
        }

        // ---- reduce the W2-dot across the 16 lanes of each half ----
#pragma unroll
        for (int m = 1; m <= 8; m <<= 1) {
#pragma unroll
            for (int v = 0; v < 8; ++v)
                outacc[v] += __shfl_xor(outacc[v], m, 32);
        }

        if ((lane & 15) == 0) {
#pragma unroll
            for (int v = 0; v < 8; ++v) {
                int eo = tile * TILE_M + half * 8 + v;  // M = v + 8*half
                if (eo < E) {
                    float o = outacc[v] + b2v;
                    out[eo] = 1.0f / (1.0f + __expf(-o));
                }
            }
        }
    }
#endif
}

extern "C" void kernel_launch(void* const* d_in, const int* in_sizes, int n_in,
                              void* d_out, int out_size, void* d_ws, size_t ws_size,
                              hipStream_t stream) {
    const float*     z   = (const float*)d_in[0];
    const long long* src = (const long long*)d_in[1];
    const long long* dst = (const long long*)d_in[2];
    const float*     W1  = (const float*)d_in[3];
    const float*     b1  = (const float*)d_in[4];
    const float*     W2  = (const float*)d_in[5];
    const float*     b2  = (const float*)d_in[6];
    float* out = (float*)d_out;

    const int E  = in_sizes[1];
    const int nZ = in_sizes[0];              // N * 256 floats

    unsigned short* Wt = (unsigned short*)d_ws;                      // 128KB
    unsigned short* Zb = (unsigned short*)((char*)d_ws + WT_BYTES);  // bf16 z

    const size_t need_zbf = (size_t)WT_BYTES + (size_t)nZ * 2;
    const bool use_zbf = (ws_size >= need_zbf) && ((nZ & 3) == 0);

    // Prep: transpose + bf16-convert W1 (65536 elems).
    edgedec_prep_w1t<<<CH, CH, 0, stream>>>(W1, Wt);

    const int nTiles = (E + TILE_M - 1) / TILE_M;
    const int blocks = (nTiles + WAVES_PER_BLOCK - 1) / WAVES_PER_BLOCK;

    if (use_zbf) {
        const int n4 = nZ >> 2;
        edgedec_prep_zbf<<<(n4 + 255) / 256, 256, 0, stream>>>(
            (const float4*)z, (ushort4*)Zb, n4);
        edgedec_main<true><<<blocks, 256, WT_BYTES, stream>>>(
            z, Zb, src, dst, Wt, b1, W2, b2, out, E);
    } else {
        edgedec_main<false><<<blocks, 256, WT_BYTES, stream>>>(
            z, Zb, src, dst, Wt, b1, W2, b2, out, E);
    }
}
